// Attention_79267916415178
// MI455X (gfx1250) — matrix operand order repaired and checked by reading, not compiled, not run on hardware
//
#include <hip/hip_runtime.h>
#include <math.h>

// Problem constants (from reference)
#define B_  128
#define S_  1024
#define E_  512
#define P_  256
#define H_  1024
#define A_  128
#define F_  32
#define KW_ 31
#define HALO_ 15
#define KX_ (P_ + E_ + H_)   // 1792 concat-K for the LSTM GEMM
#define N4H_ (4 * H_)        // 4096

typedef __attribute__((ext_vector_type(16))) _Float16 v16h;
typedef __attribute__((ext_vector_type(8)))  _Float16 h8;     // one 16B (b128) load
typedef __attribute__((ext_vector_type(8)))  float    v8f;

union Frag16 { v16h v; h8 q[2]; _Float16 h[16]; };

__device__ __forceinline__ float sigmoidf_(float x) { return 1.0f / (1.0f + __expf(-x)); }

// ---------------------------------------------------------------------------
// Async global->LDS staging (CDNA5 ASYNCcnt path) with portable fallback.
// Flat pointers: global VA == flat VA; LDS offset == low 32 bits of flat VA
// (ISA §10.2 aperture mapping), so uintptr_t round-trips are valid.
// The builtin takes typed AS1/AS3 int* operands.
// ---------------------------------------------------------------------------
#if defined(__has_builtin)
#if __has_builtin(__builtin_amdgcn_global_load_async_to_lds_b32)
#define HAVE_ASYNC_LDS 1
#endif
#endif

typedef __attribute__((address_space(1))) int g1_int;
typedef __attribute__((address_space(3))) int l3_int;

__device__ __forceinline__ void lds_fill_async(float* dst_lds, const float* src,
                                               int count, int tid, int nthreads) {
#ifdef HAVE_ASYNC_LDS
  for (int i = tid; i < count; i += nthreads) {
    __builtin_amdgcn_global_load_async_to_lds_b32(
        (g1_int*)(uintptr_t)(src + i),
        (l3_int*)(uint32_t)(uintptr_t)(dst_lds + i),
        0, 0);
  }
#else
  for (int i = tid; i < count; i += nthreads) dst_lds[i] = src[i];
#endif
}

__device__ __forceinline__ void lds_fill_wait() {
#ifdef HAVE_ASYNC_LDS
#if __has_builtin(__builtin_amdgcn_s_wait_asynccnt)
  __builtin_amdgcn_s_wait_asynccnt(0);
#else
  asm volatile("s_wait_asynccnt 0" ::: "memory");
#endif
#endif
}

// ---------------------------------------------------------------------------
// Pack kernels: build f16 K-major operands once per launch (~45MB of traffic,
// ~2us at 23.3 TB/s; removes every branch/convert from the WMMA inner loop).
// ---------------------------------------------------------------------------
__global__ void pack_x16_kernel(const float* __restrict__ prenet,
                                const float* __restrict__ prev_ctx,
                                const float* __restrict__ att_h,
                                _Float16* __restrict__ X16) {
  int idx = blockIdx.x * blockDim.x + threadIdx.x;   // 0 .. B*KX_-1
  int b = idx / KX_;
  int k = idx - b * KX_;
  float v;
  if (k < P_)            v = prenet[b * P_ + k];
  else if (k < P_ + E_)  v = prev_ctx[b * E_ + (k - P_)];
  else                   v = att_h[b * H_ + (k - P_ - E_)];
  X16[idx] = (_Float16)v;
}

__global__ void pack_w16_kernel(const float* __restrict__ W_ih,
                                const float* __restrict__ W_hh,
                                _Float16* __restrict__ W16) {
  size_t idx = (size_t)blockIdx.x * blockDim.x + threadIdx.x; // 0 .. 4096*1792-1
  int n = (int)(idx / KX_);
  int k = (int)(idx - (size_t)n * KX_);
  float v = (k < P_ + E_) ? W_ih[(size_t)n * (P_ + E_) + k]
                          : W_hh[(size_t)n * H_ + (k - P_ - E_)];
  W16[idx] = (_Float16)v;
}

__global__ void cvt_f16_kernel(const float* __restrict__ src,
                               _Float16* __restrict__ dst) {
  int idx = blockIdx.x * blockDim.x + threadIdx.x;
  dst[idx] = (_Float16)src[idx];
}

// ---------------------------------------------------------------------------
// WMMA GEMM, f16 inputs / f32 accumulate:  C[M,N] = A16[M,K] * B16[N,K]^T (+bias)
// Both operands K-major f16. One wave computes a 16(M) x 64(N) tile.
// This code uses K-contiguous per-lane fragment data, same order for A and B (not the ISA table order):
//   A lane(m=lane&15, kh=lane>>4): halves [kh*16, kh*16+16)
//   B lane(n=lane&15, kh=lane>>4): halves [kh*16, kh*16+16)
// -> 2 (A) + 2 (B) global_load_b128 per fragment, no scalar gathers.
// blockDim.x = 128 (4 waves); wave-uniform early-out keeps EXEC all-ones.
// ---------------------------------------------------------------------------
__global__ void wmma_gemm_f16(const _Float16* __restrict__ A16, int lda,
                              const _Float16* __restrict__ B16, int ldb,
                              const float* __restrict__ bias,
                              float* __restrict__ Cmat,
                              int N, int Ktot) {
  const int lane = threadIdx.x & 31;
  const int wave = threadIdx.x >> 5;
  const int nrow = lane & 15;   // A: M row | B/C: N column
  const int kh   = lane >> 4;   // K-half selector
  const int m0   = blockIdx.y * 16;
  const int n0   = (blockIdx.x * 4 + wave) * 64;
  if (n0 >= N) return;          // wave-uniform

  const _Float16* arow  = &A16[(size_t)(m0 + nrow) * lda + kh * 16];
  const _Float16* brow  = &B16[(size_t)(n0 + nrow) * ldb + kh * 16];
  const size_t bstep = (size_t)16 * ldb;   // next 16-column group

  v8f acc0 = {}, acc1 = {}, acc2 = {}, acc3 = {};

  for (int k0 = 0; k0 < Ktot; k0 += 32) {
    Frag16 a;
    a.q[0] = *(const h8*)&arow[k0];        // K = kh*16    .. +8
    a.q[1] = *(const h8*)&arow[k0 + 8];    // K = kh*16+8  .. +8

    Frag16 b0, b1, b2, b3;
    b0.q[0] = *(const h8*)&brow[k0];
    b0.q[1] = *(const h8*)&brow[k0 + 8];
    b1.q[0] = *(const h8*)&brow[bstep + k0];
    b1.q[1] = *(const h8*)&brow[bstep + k0 + 8];
    b2.q[0] = *(const h8*)&brow[2 * bstep + k0];
    b2.q[1] = *(const h8*)&brow[2 * bstep + k0 + 8];
    b3.q[0] = *(const h8*)&brow[3 * bstep + k0];
    b3.q[1] = *(const h8*)&brow[3 * bstep + k0 + 8];

    acc0 = __builtin_amdgcn_wmma_f32_16x16x32_f16(false, a.v, false, b0.v, (short)0, acc0, false, false);
    acc1 = __builtin_amdgcn_wmma_f32_16x16x32_f16(false, a.v, false, b1.v, (short)0, acc1, false, false);
    acc2 = __builtin_amdgcn_wmma_f32_16x16x32_f16(false, a.v, false, b2.v, (short)0, acc2, false, false);
    acc3 = __builtin_amdgcn_wmma_f32_16x16x32_f16(false, a.v, false, b3.v, (short)0, acc3, false, false);
  }

  // C 16x16 f32 layout: lane n = nrow, VGPR r -> M = r + 8*kh
#pragma unroll
  for (int nt = 0; nt < 4; ++nt) {
    const v8f acc = (nt == 0) ? acc0 : (nt == 1) ? acc1 : (nt == 2) ? acc2 : acc3;
    int col = n0 + nt * 16 + nrow;
    float badd = bias ? bias[col] : 0.0f;
#pragma unroll
    for (int r = 0; r < 8; ++r) {
      int m = m0 + r + 8 * kh;
      Cmat[(size_t)m * N + col] = acc[r] + badd;
    }
  }
}

// ---------------------------------------------------------------------------
// LSTM gate activations. gates[B,4H] (biases folded here) -> h16[B,H] (f16 for
// the query WMMA GEMM). torch gate order i, f, g, o in column blocks of H.
// ---------------------------------------------------------------------------
__global__ void lstm_act_kernel(const float* __restrict__ gates,
                                const float* __restrict__ att_c,
                                const float* __restrict__ b_ih,
                                const float* __restrict__ b_hh,
                                _Float16* __restrict__ h16) {
  int idx = blockIdx.x * blockDim.x + threadIdx.x;  // b*H + h
  int b  = idx >> 10;
  int hh = idx & (H_ - 1);
  const float* g = &gates[(size_t)b * N4H_];
  float gi = g[hh]          + b_ih[hh]          + b_hh[hh];
  float gf = g[H_ + hh]     + b_ih[H_ + hh]     + b_hh[H_ + hh];
  float gg = g[2 * H_ + hh] + b_ih[2 * H_ + hh] + b_hh[2 * H_ + hh];
  float go = g[3 * H_ + hh] + b_ih[3 * H_ + hh] + b_hh[3 * H_ + hh];
  float c  = sigmoidf_(gf) * att_c[idx] + sigmoidf_(gi) * tanhf(gg);
  h16[idx] = (_Float16)(sigmoidf_(go) * tanhf(c));
}

// ---------------------------------------------------------------------------
// Fused location conv + W_loc projection + tanh score.
// One block = 256 consecutive s of one batch row; streams proc_mem once,
// loc[B,S,A] (64MB) never touches HBM. Big LDS fills use the async path.
// ---------------------------------------------------------------------------
__global__ void loc_score_kernel(const float* __restrict__ cum_w,
                                 const float* __restrict__ prev_w,
                                 const float* __restrict__ proc_mem,
                                 const float* __restrict__ conv_w,
                                 const float* __restrict__ conv_b,
                                 const float* __restrict__ W_loc,
                                 const float* __restrict__ b_loc,
                                 const float* __restrict__ W_out,
                                 const float* __restrict__ b_out,
                                 const float* __restrict__ qry,
                                 const float* __restrict__ mask,
                                 float* __restrict__ scores) {
  const int tid = threadIdx.x;          // 0..255
  const int b   = blockIdx.y;
  const int s0  = blockIdx.x * 256;

  __shared__ float s_convw[F_ * 2 * KW_];   // 1984
  __shared__ float s_convb[F_];
  __shared__ float s_Wloc[A_ * F_];         // 4096
  __shared__ float s_bloc[A_];
  __shared__ float s_Wout[A_];
  __shared__ float s_qry[A_];
  __shared__ float s_cum[256 + 2 * HALO_];
  __shared__ float s_prev[256 + 2 * HALO_];

  lds_fill_async(s_convw, conv_w, F_ * 2 * KW_, tid, 256);
  lds_fill_async(s_Wloc,  W_loc,  A_ * F_,      tid, 256);
  if (tid < F_) s_convb[tid] = conv_b[tid];
  if (tid < A_) {
    s_bloc[tid] = b_loc[tid];
    s_Wout[tid] = W_out[tid];
    s_qry[tid]  = qry[(size_t)b * A_ + tid];
  }
  for (int i = tid; i < 256 + 2 * HALO_; i += 256) {
    int sg = s0 + i - HALO_;
    bool ok = (sg >= 0) && (sg < S_);
    s_cum[i]  = ok ? cum_w[(size_t)b * S_ + sg]  : 0.0f;
    s_prev[i] = ok ? prev_w[(size_t)b * S_ + sg] : 0.0f;
  }
  lds_fill_wait();
  __syncthreads();

  const int s = s0 + tid;

  // Conv1d (cross-correlation, zero-pad 15): locf[f] for this position
  float locf[F_];
#pragma unroll 4
  for (int f = 0; f < F_; ++f) {
    float acc = s_convb[f];
    const float* w0 = &s_convw[(f * 2 + 0) * KW_];
    const float* w1 = &s_convw[(f * 2 + 1) * KW_];
#pragma unroll
    for (int k = 0; k < KW_; ++k)
      acc += w0[k] * s_cum[tid + k] + w1[k] * s_prev[tid + k];
    locf[f] = acc;
  }

  // score = sum_a W_out[a] * tanh(qry[a] + proc_mem[b,s,a] + (W_loc[a,:]·locf + b_loc[a]))
  const float* pm = &proc_mem[((size_t)b * S_ + s) * A_];
  float sc = 0.0f;
  for (int a = 0; a < A_; ++a) {
    const float* wl = &s_Wloc[a * F_];
    float la = s_bloc[a];
#pragma unroll
    for (int f = 0; f < F_; ++f) la += wl[f] * locf[f];
    float t = tanhf(s_qry[a] + pm[a] + la);
    sc += s_Wout[a] * t;
  }
  scores[(size_t)b * S_ + s] = sc + b_out[0] + mask[(size_t)b * S_ + s];
}

// ---------------------------------------------------------------------------
// Softmax over S=1024 per row. One block (256 threads) per b.
// ---------------------------------------------------------------------------
__global__ void softmax_kernel(const float* __restrict__ scores,
                               float* __restrict__ weights) {
  const int b = blockIdx.x, tid = threadIdx.x;
  __shared__ float red[256];
  const float* row = &scores[(size_t)b * S_];
  float v[4];
  float m = -INFINITY;
#pragma unroll
  for (int i = 0; i < 4; ++i) { v[i] = row[tid + 256 * i]; m = fmaxf(m, v[i]); }
  red[tid] = m; __syncthreads();
  for (int off = 128; off > 0; off >>= 1) {
    if (tid < off) red[tid] = fmaxf(red[tid], red[tid + off]);
    __syncthreads();
  }
  const float mx = red[0]; __syncthreads();
  float sum = 0.0f;
#pragma unroll
  for (int i = 0; i < 4; ++i) { v[i] = __expf(v[i] - mx); sum += v[i]; }
  red[tid] = sum; __syncthreads();
  for (int off = 128; off > 0; off >>= 1) {
    if (tid < off) red[tid] += red[tid + off];
    __syncthreads();
  }
  const float inv = 1.0f / red[0];
#pragma unroll
  for (int i = 0; i < 4; ++i) weights[(size_t)b * S_ + tid + 256 * i] = v[i] * inv;
}

// ---------------------------------------------------------------------------
// context[b,e] = sum_s w[b,s] * enc[b,s,e]. Streams enc_seq (256 MB, the
// roofline term) exactly once, coalesced over e, with global prefetch ahead.
// ---------------------------------------------------------------------------
__global__ void context_kernel(const float* __restrict__ weights,
                               const float* __restrict__ enc,
                               float* __restrict__ out) {
  const int b = blockIdx.y;
  const int e = blockIdx.x * 256 + threadIdx.x;
  __shared__ float sw[S_];
  lds_fill_async(sw, &weights[(size_t)b * S_], S_, threadIdx.x, 256);
  lds_fill_wait();
  __syncthreads();

  const float* ep = &enc[((size_t)b * S_) * E_ + e];
  float acc = 0.0f;
  for (int s = 0; s < S_; s += 4) {
    __builtin_prefetch(&ep[(size_t)(s + 16) * E_], 0, 0);  // global_prefetch_b8
    acc += sw[s]     * ep[(size_t)s * E_]
         + sw[s + 1] * ep[(size_t)(s + 1) * E_]
         + sw[s + 2] * ep[(size_t)(s + 2) * E_]
         + sw[s + 3] * ep[(size_t)(s + 3) * E_];
  }
  out[(size_t)b * E_ + e] = acc;
}

// ---------------------------------------------------------------------------
// Launch
// ---------------------------------------------------------------------------
extern "C" void kernel_launch(void* const* d_in, const int* in_sizes, int n_in,
                              void* d_out, int out_size, void* d_ws, size_t ws_size,
                              hipStream_t stream) {
  const float* prenet   = (const float*)d_in[0];
  const float* prev_ctx = (const float*)d_in[1];
  const float* att_h    = (const float*)d_in[2];
  const float* att_c    = (const float*)d_in[3];
  const float* prev_w   = (const float*)d_in[4];
  const float* cum_w    = (const float*)d_in[5];
  const float* enc_seq  = (const float*)d_in[6];
  const float* proc_mem = (const float*)d_in[7];
  const float* mask     = (const float*)d_in[8];
  const float* W_ih     = (const float*)d_in[9];
  const float* W_hh     = (const float*)d_in[10];
  const float* b_ih     = (const float*)d_in[11];
  const float* b_hh     = (const float*)d_in[12];
  const float* conv_w   = (const float*)d_in[13];
  const float* conv_b   = (const float*)d_in[14];
  const float* W_loc    = (const float*)d_in[15];
  const float* b_loc    = (const float*)d_in[16];
  const float* W_q      = (const float*)d_in[17];
  const float* b_q      = (const float*)d_in[18];
  const float* W_out    = (const float*)d_in[19];
  const float* b_out    = (const float*)d_in[20];
  float* out = (float*)d_out;

  // Workspace layout (~19 MB)
  char* wsb = (char*)d_ws;
  _Float16* X16   = (_Float16*)wsb;                       wsb += (size_t)B_ * KX_ * 2;       // 448 KB
  _Float16* W16   = (_Float16*)wsb;                       wsb += (size_t)N4H_ * KX_ * 2;     // 14.0 MB
  _Float16* Wq16  = (_Float16*)wsb;                       wsb += (size_t)A_ * H_ * 2;        // 256 KB
  _Float16* h16   = (_Float16*)wsb;                       wsb += (size_t)B_ * H_ * 2;        // 256 KB
  float*    gates = (float*)wsb;                          wsb += (size_t)B_ * N4H_ * 4;      // 2 MB
  float*    qry   = (float*)wsb;                          wsb += (size_t)B_ * A_ * 4;        // 64 KB
  float*    scores= (float*)wsb;                          wsb += (size_t)B_ * S_ * 4;        // 512 KB
  float*    wts   = (float*)wsb;                          wsb += (size_t)B_ * S_ * 4;        // 512 KB

  // 0) pack f16 operands
  pack_x16_kernel<<<(B_ * KX_) / 256, 256, 0, stream>>>(prenet, prev_ctx, att_h, X16);
  pack_w16_kernel<<<(int)(((size_t)N4H_ * KX_) / 256), 256, 0, stream>>>(W_ih, W_hh, W16);
  cvt_f16_kernel<<<(A_ * H_) / 256, 256, 0, stream>>>(W_q, Wq16);

  // 1) gates = X @ [W_ih | W_hh]^T   (WMMA f16 in / f32 acc)  M=128, N=4096, K=1792
  wmma_gemm_f16<<<dim3(N4H_ / 256, B_ / 16), 128, 0, stream>>>(
      X16, KX_, W16, KX_, /*bias=*/nullptr, gates, N4H_, KX_);

  // 2) LSTM activations -> h16
  lstm_act_kernel<<<(B_ * H_) / 256, 256, 0, stream>>>(gates, att_c, b_ih, b_hh, h16);

  // 3) qry = h @ W_q^T + b_q   (WMMA)  M=128, N=128, K=1024
  wmma_gemm_f16<<<dim3(1, B_ / 16), 128, 0, stream>>>(
      h16, H_, Wq16, H_, /*bias=*/b_q, qry, A_, H_);

  // 4) fused conv + loc-projection + tanh score (streams proc_mem once)
  loc_score_kernel<<<dim3(S_ / 256, B_), 256, 0, stream>>>(
      cum_w, prev_w, proc_mem, conv_w, conv_b, W_loc, b_loc, W_out, b_out,
      qry, mask, scores);

  // 5) softmax over S
  softmax_kernel<<<B_, 256, 0, stream>>>(scores, wts);

  // 6) context = weights @ enc_seq (streams enc_seq once, prefetched)
  context_kernel<<<dim3(E_ / 256, B_), 256, 0, stream>>>(wts, enc_seq, out);
}
